// MemRNN_4887672783623
// MI455X (gfx1250) — matrix operand order repaired and checked by reading, not compiled, not run on hardware
//
#include <hip/hip_runtime.h>
#include <math.h>

// ---------------------------------------------------------------------------
// MemRNN forward for MI455X (gfx1250, wave32, WMMA).
// T=256, B=64, H=512, INP=512.  All GEMMs via v_wmma_f32_16x16x32_bf16 with
// the A panel staged through LDS using async global->LDS copies (ASYNCcnt).
// d_out (f32 [T,B,H]) doubles as the "mem" buffer (out[t] == mem[t]).
// ---------------------------------------------------------------------------

#define T_DIM 256
#define B_DIM 64
#define H_DIM 512
#define I_DIM 512

typedef __attribute__((ext_vector_type(16))) __bf16 bfrag16;
typedef __attribute__((ext_vector_type(8)))  __bf16 bfrag8;
typedef __attribute__((ext_vector_type(8)))  float  v8f;

// B-matrix 32x16 bf16 fragment: lane group g=l>>4 holds 16 contiguous K
// values at kk + g*16 (column n = l&15).  One 32-byte load.
__device__ __forceinline__ bfrag16 load_fragB(const __bf16* p) {
    return *reinterpret_cast<const bfrag16*>(p);
}

// ---------------------------------------------------------------------------
// Cooperative GEMM:  out[m,n] = epi( sum_k A[m,k]*W[n,k] (+addend) (+bias) )
// A: [M,K] bf16 row-major, W: [N,K] bf16 row-major  (computes A @ W^T).
// One block (8 waves) -> 16 x 512 output tile; wave w owns columns [w*64,
// w*64+64).  The shared 16-row A panel is staged chunk-by-chunk (16x32 bf16 =
// 1 KB) into double-buffered LDS with global_load_async_to_lds_b128; all 8
// waves then read A fragments from LDS (ds_load_b128) and B fragments
// straight from L2.
// ---------------------------------------------------------------------------
__global__ __launch_bounds__(256)
void gemm_bf16_wmma(const __bf16* __restrict__ A,
                    const __bf16* __restrict__ W,
                    const float*  __restrict__ addend,
                    const float*  __restrict__ bias,
                    float*        __restrict__ out,
                    __bf16*       __restrict__ out_bf,
                    int N, int K, int apply_tanh)
{
    __shared__ __bf16 shA[2][16 * 32];   // double-buffered 16x32 A chunk

    const int lane = threadIdx.x & 31;
    const int wv   = threadIdx.x >> 5;
    const int m0   = blockIdx.x << 4;
    const int n0   = wv << 6;
    const int rc   = lane & 15;
    const int g8   = (lane >> 4) << 3;   // unused; A frag uses g16 below
    const int g16  = (lane >> 4) << 4;   // B-frag K sub-chunk base

    // --- async staging setup (wave 0): two b128 issues move 16 rows x 64 B
    const int seg  = lane & 3;           // 16-byte segment within a row
    const int row0 = lane >> 2;          // rows 0..7 (instr0) / 8..15 (instr1)
    const __bf16* gA0 = A + (size_t)(m0 + row0)     * K + seg * 8;
    const __bf16* gA1 = A + (size_t)(m0 + 8 + row0) * K + seg * 8;
    const unsigned lds0 =
        (unsigned)(unsigned long long)&shA[0][row0 * 32 + seg * 8];
    const unsigned lds1 =
        (unsigned)(unsigned long long)&shA[0][(8 + row0) * 32 + seg * 8];
    const unsigned bufb = (unsigned)(16 * 32 * sizeof(__bf16));  // 1024 B

    if (wv == 0) {  // prologue: stage chunk 0 into buffer 0
        asm volatile("global_load_async_to_lds_b128 %0, %1, off"
                     :: "v"(lds0), "v"((const void*)gA0) : "memory");
        asm volatile("global_load_async_to_lds_b128 %0, %1, off"
                     :: "v"(lds1), "v"((const void*)gA1) : "memory");
    }

    const __bf16* b0p = W + (size_t)(n0 +  0 + rc) * K + g16;
    const __bf16* b1p = W + (size_t)(n0 + 16 + rc) * K + g16;
    const __bf16* b2p = W + (size_t)(n0 + 32 + rc) * K + g16;
    const __bf16* b3p = W + (size_t)(n0 + 48 + rc) * K + g16;

    v8f acc0 = {}, acc1 = {}, acc2 = {}, acc3 = {};
    unsigned buf = 0;
    for (int kk = 0; kk < K; kk += 32) {
        if (wv == 0) {
            if (kk + 32 < K) {           // stage next chunk into other buffer
                unsigned nb = (buf ^ 1u) * bufb;
                asm volatile("global_load_async_to_lds_b128 %0, %1, off"
                             :: "v"(lds0 + nb),
                                "v"((const void*)(gA0 + kk + 32)) : "memory");
                asm volatile("global_load_async_to_lds_b128 %0, %1, off"
                             :: "v"(lds1 + nb),
                                "v"((const void*)(gA1 + kk + 32)) : "memory");
                // current chunk done (async loads complete in order);
                // next chunk (2 transfers) may remain in flight
                asm volatile("s_wait_asynccnt 0x2" ::: "memory");
            } else {
                asm volatile("s_wait_asynccnt 0x0" ::: "memory");
            }
        }
        __syncthreads();                 // A chunk visible to all waves

        // A fragment from LDS (16 contiguous K at +g16, matching B)
        const bfrag8* qa =
            reinterpret_cast<const bfrag8*>(&shA[buf][rc * 32 + g16]);
        bfrag8 lo = qa[0];
        bfrag8 hi = qa[1];
        bfrag16 a;
#pragma unroll
        for (int j = 0; j < 8; ++j) { a[j] = lo[j]; a[8 + j] = hi[j]; }

        bfrag16 b0 = load_fragB(b0p + kk);
        bfrag16 b1 = load_fragB(b1p + kk);
        bfrag16 b2 = load_fragB(b2p + kk);
        bfrag16 b3 = load_fragB(b3p + kk);
        acc0 = __builtin_amdgcn_wmma_f32_16x16x32_bf16(false, a, false, b0, (short)0, acc0, false, false);
        acc1 = __builtin_amdgcn_wmma_f32_16x16x32_bf16(false, a, false, b1, (short)0, acc1, false, false);
        acc2 = __builtin_amdgcn_wmma_f32_16x16x32_bf16(false, a, false, b2, (short)0, acc2, false, false);
        acc3 = __builtin_amdgcn_wmma_f32_16x16x32_bf16(false, a, false, b3, (short)0, acc3, false, false);

        __syncthreads();                 // reads done before buffer reuse
        buf ^= 1u;
    }

    const int mrow = m0 + ((lane >> 4) << 3);
#pragma unroll
    for (int r = 0; r < 8; ++r) {
        const int m = mrow + r;
#pragma unroll
        for (int j = 0; j < 4; ++j) {
            const int n = n0 + (j << 4) + rc;
            float x = (j == 0) ? acc0[r] : (j == 1) ? acc1[r]
                    : (j == 2) ? acc2[r] : acc3[r];
            const size_t idx = (size_t)m * N + n;
            if (addend) x += addend[idx];
            if (bias)   x += bias[n];
            if (apply_tanh) x = tanhf(x);
            out[idx] = x;
            if (out_bf) out_bf[idx] = (__bf16)x;
        }
    }
}

// ---------------------------------------------------------------------------
// f32 -> bf16 conversion (grid-stride)
// ---------------------------------------------------------------------------
__global__ void f32_to_bf16(const float* __restrict__ src,
                            __bf16* __restrict__ dst, int n)
{
    int i = blockIdx.x * blockDim.x + threadIdx.x;
    int stride = gridDim.x * blockDim.x;
    for (; i < n; i += stride) dst[i] = (__bf16)src[i];
}

// ---------------------------------------------------------------------------
// Step 0: h0_pre = xU[0] (bias already included); h0 = tanh(h0_pre)
// out[0] = h0 (mem[0]); st_bf = bf16(h0_pre); h_bf = bf16(h0)
// ---------------------------------------------------------------------------
__global__ void step0_kernel(const float* __restrict__ xU0,
                             float* __restrict__ out0,
                             __bf16* __restrict__ st_bf,
                             __bf16* __restrict__ h_bf, int n)
{
    int i = blockIdx.x * blockDim.x + threadIdx.x;
    if (i >= n) return;
    float pre = xU0[i];
    float h   = tanhf(pre);
    out0[i]  = h;
    st_bf[i] = (__bf16)pre;
    h_bf[i]  = (__bf16)h;
}

// ---------------------------------------------------------------------------
// Attention for step t: one block per batch element b (256 threads = 8 waves).
//   e[tau] = sum_h v[h]*tanh(stVa[b,h] + Uam[tau,b,h]),  tau < t
//   alphas = softmax_tau(e);  ct = sum_tau alphas[tau]*mem[tau,b,:]
//   st_new = mem[t-1,b,:] + ct    -> written as bf16 into st_bf
// ---------------------------------------------------------------------------
__global__ void attn_step_kernel(const float* __restrict__ stVa,
                                 const float* __restrict__ Uam,
                                 const float* __restrict__ mem,
                                 const float* __restrict__ v,
                                 __bf16* __restrict__ st_bf, int t)
{
    __shared__ float sh_e[T_DIM];
    __shared__ float sh_sum;

    const int b    = blockIdx.x;
    const int tid  = threadIdx.x;
    const int lane = tid & 31;
    const int wv   = tid >> 5;
    const size_t bh = (size_t)b * H_DIM;

    // phase 1: logits, wave-parallel over tau, lane-parallel over h
    for (int tau = wv; tau < t; tau += 8) {
        const float* um = Uam + (size_t)tau * B_DIM * H_DIM + bh;
        float s = 0.f;
        for (int h = lane; h < H_DIM; h += 32)
            s += v[h] * tanhf(stVa[bh + h] + um[h]);
#pragma unroll
        for (int off = 16; off; off >>= 1) s += __shfl_xor(s, off, 32);
        if (lane == 0) sh_e[tau] = s;
    }
    __syncthreads();

    // phase 2: softmax stats (wave 0), sh_e[tau] <- exp(e - max)
    if (wv == 0) {
        float mx = -3.4e38f;
        for (int tau = lane; tau < t; tau += 32) mx = fmaxf(mx, sh_e[tau]);
#pragma unroll
        for (int off = 16; off; off >>= 1) mx = fmaxf(mx, __shfl_xor(mx, off, 32));
        float sum = 0.f;
        for (int tau = lane; tau < t; tau += 32) {
            float p = __expf(sh_e[tau] - mx);
            sh_e[tau] = p;
            sum += p;
        }
#pragma unroll
        for (int off = 16; off; off >>= 1) sum += __shfl_xor(sum, off, 32);
        if (lane == 0) sh_sum = sum;
    }
    __syncthreads();
    const float inv = 1.0f / sh_sum;

    // phase 3: ct and st_new (threads over h)
    for (int h = tid; h < H_DIM; h += blockDim.x) {
        float acc = 0.f;
        for (int tau = 0; tau < t; ++tau)
            acc += sh_e[tau] * mem[(size_t)tau * B_DIM * H_DIM + bh + h];
        float stn = mem[(size_t)(t - 1) * B_DIM * H_DIM + bh + h] + acc * inv;
        st_bf[bh + h] = (__bf16)stn;
    }
}

// ---------------------------------------------------------------------------
extern "C" void kernel_launch(void* const* d_in, const int* in_sizes, int n_in,
                              void* d_out, int out_size, void* d_ws, size_t ws_size,
                              hipStream_t stream)
{
    (void)in_sizes; (void)n_in; (void)out_size; (void)ws_size;

    const float* x   = (const float*)d_in[0];   // [T,B,INP]
    const float* Uw  = (const float*)d_in[1];   // [H,INP]
    const float* Ub  = (const float*)d_in[2];   // [H]
    const float* Vw  = (const float*)d_in[3];   // [H,H]
    const float* Uaw = (const float*)d_in[4];   // [H,H]
    const float* Vaw = (const float*)d_in[5];   // [H,H]
    const float* v   = (const float*)d_in[6];   // [1,H]
    float* out = (float*)d_out;                 // [T,B,H]  == mem

    // workspace carve-up (256B aligned)
    char*  ws  = (char*)d_ws;
    size_t off = 0;
    auto carve = [&](size_t bytes) -> char* {
        char* p = ws + off;
        off = (off + bytes + 255) & ~(size_t)255;
        return p;
    };
    const size_t TBH = (size_t)T_DIM * B_DIM * H_DIM;
    const size_t BH  = (size_t)B_DIM * H_DIM;
    float*  xU     = (float*)carve(TBH * 4);
    float*  Uam    = (float*)carve(TBH * 4);
    float*  stVa   = (float*)carve(BH * 4);
    __bf16* x_bf   = (__bf16*)carve((size_t)T_DIM * B_DIM * I_DIM * 2);
    __bf16* Uw_bf  = (__bf16*)carve((size_t)H_DIM * I_DIM * 2);
    __bf16* Vw_bf  = (__bf16*)carve((size_t)H_DIM * H_DIM * 2);
    __bf16* Uaw_bf = (__bf16*)carve((size_t)H_DIM * H_DIM * 2);
    __bf16* Vaw_bf = (__bf16*)carve((size_t)H_DIM * H_DIM * 2);
    __bf16* st_bf  = (__bf16*)carve(BH * 2);
    __bf16* h_bf   = (__bf16*)carve(BH * 2);

    auto cvt = [&](const float* s, __bf16* d, int n) {
        int blocks = (n + 255) / 256; if (blocks > 4096) blocks = 4096;
        f32_to_bf16<<<blocks, 256, 0, stream>>>(s, d, n);
    };
    auto gemm = [&](const __bf16* A, const __bf16* W, const float* add,
                    const float* bias, float* o, __bf16* obf,
                    int M, int N, int K, int do_tanh) {
        gemm_bf16_wmma<<<M / 16, 256, 0, stream>>>(A, W, add, bias, o, obf,
                                                   N, K, do_tanh);
    };

    // 1) convert inputs/weights to bf16
    cvt(x,   x_bf,   T_DIM * B_DIM * I_DIM);
    cvt(Uw,  Uw_bf,  H_DIM * I_DIM);
    cvt(Vw,  Vw_bf,  H_DIM * H_DIM);
    cvt(Uaw, Uaw_bf, H_DIM * H_DIM);
    cvt(Vaw, Vaw_bf, H_DIM * H_DIM);

    // 2) xU[t,b,:] = x[t,b,:] @ Uw^T + Ub   (one big WMMA GEMM, M = T*B)
    gemm(x_bf, Uw_bf, nullptr, Ub, xU, nullptr,
         T_DIM * B_DIM, H_DIM, I_DIM, 0);

    // 3) step 0: h0 = tanh(h0_pre); mem[0]=h0; st = h0_pre
    step0_kernel<<<(int)((BH + 255) / 256), 256, 0, stream>>>(
        xU, out, st_bf, h_bf, (int)BH);

    // 4) Uam[0] = h0 @ Uaw^T
    gemm(h_bf, Uaw_bf, nullptr, nullptr, Uam, nullptr, B_DIM, H_DIM, H_DIM, 0);

    // 5) sequential steps
    for (int t = 1; t < T_DIM; ++t) {
        // stVa = st @ Vaw^T
        gemm(st_bf, Vaw_bf, nullptr, nullptr, stVa, nullptr,
             B_DIM, H_DIM, H_DIM, 0);
        // attention -> st_bf := bf16(st_new)
        attn_step_kernel<<<B_DIM, 256, 0, stream>>>(stVa, Uam, out, v, st_bf, t);
        // h = tanh(xU[t] + st_new @ Vw^T); mem[t] = h; h_bf = bf16(h)
        gemm(st_bf, Vw_bf, xU + (size_t)t * BH, nullptr,
             out + (size_t)t * BH, h_bf, B_DIM, H_DIM, H_DIM, 1);
        // Uam[t] = h @ Uaw^T
        gemm(h_bf, Uaw_bf, nullptr, nullptr, Uam + (size_t)t * BH, nullptr,
             B_DIM, H_DIM, H_DIM, 0);
    }
}
